// Word2ManSkipGramOneHot_40097814675558
// MI455X (gfx1250) — compile-verified, matrix operand order checked by reading, not run on hardware
//
#include <hip/hip_runtime.h>
#include <hip/hip_bf16.h>

// Problem constants (from reference)
#define BB 1024
#define VV 50257
#define EE 128

typedef _Float16 v8h  __attribute__((ext_vector_type(8)));
typedef _Float16 v16h __attribute__((ext_vector_type(16)));
typedef float    v8f  __attribute__((ext_vector_type(8)));

// ---------------------------------------------------------------------------
// K1: recover indices from one-hot rows. One block per row, strided scan.
// ---------------------------------------------------------------------------
__global__ __launch_bounds__(256) void find_idx_kernel(const float* __restrict__ xs,
                                                       int* __restrict__ idx) {
    const size_t b = blockIdx.x;
    const float* row = xs + b * (size_t)VV;
    for (int v = threadIdx.x; v < VV; v += 256) {
        if (row[v] > 0.5f) idx[b] = v;   // exactly one lane matches per row
    }
}

// ---------------------------------------------------------------------------
// K2: convert EMBEDM f32 -> f16 (vectorized: float4 load, 4xhalf store)
// ---------------------------------------------------------------------------
__global__ __launch_bounds__(256) void cvt_embed_kernel(const float* __restrict__ src,
                                                        _Float16* __restrict__ dst) {
    const size_t n4 = (size_t)VV * EE / 4;
    size_t i = (size_t)blockIdx.x * 256 + threadIdx.x;
    if (i >= n4) return;
    const float4 f = reinterpret_cast<const float4*>(src)[i];
    v8h* d8 = nullptr; (void)d8;
    _Float16 h[4] = { (_Float16)f.x, (_Float16)f.y, (_Float16)f.z, (_Float16)f.w };
    // 8-byte store
    *reinterpret_cast<uint2*>(dst + i * 4) = *reinterpret_cast<uint2*>(h);
}

// ---------------------------------------------------------------------------
// K3: P[b] = EMBEDM[idx[b]] @ metric  (f32 compute, f16 store)
// One block (128 threads) per batch row; metric stays L2/WGP$-hot.
// ---------------------------------------------------------------------------
__global__ __launch_bounds__(128) void gather_project_kernel(const float* __restrict__ EM,
                                                             const float* __restrict__ metric,
                                                             const int* __restrict__ idx,
                                                             _Float16* __restrict__ Ph) {
    __shared__ float emb[EE];
    const int b = blockIdx.x;
    const int t = threadIdx.x;
    const int id = idx[b];
    emb[t] = EM[(size_t)id * EE + t];
    __syncthreads();
    float acc = 0.f;
    #pragma unroll 8
    for (int e1 = 0; e1 < EE; ++e1) {
        acc = fmaf(emb[e1], metric[e1 * EE + t], acc);   // coalesced over t
    }
    Ph[(size_t)b * EE + t] = (_Float16)acc;
}

// ---------------------------------------------------------------------------
// K4: scores = P @ EMBEDM_f16^T via v_wmma_f32_16x16x32_f16.
// Fragment loader for a row-major [16 x 128] tile (rows contiguous, K inner).
// CDNA5 16-bit A layout: lanes 0-15 hold K={0..7,16..23}, lanes 16-31 hold
// K={8..15,24..31} per 32-wide K step; B (columns = EMBEDM rows) mirrors it.
// ---------------------------------------------------------------------------
__device__ __forceinline__ v16h load_frag16(const _Float16* __restrict__ rowp,
                                            int k0, int hi) {
    const v8h lo = *reinterpret_cast<const v8h*>(rowp + k0 + 8 * hi);       // K k0+8hi ..
    const v8h up = *reinterpret_cast<const v8h*>(rowp + k0 + 16 + 8 * hi);  // K k0+16+8hi ..
    return __builtin_shufflevector(lo, up, 0,1,2,3,4,5,6,7,8,9,10,11,12,13,14,15);
}

__global__ __launch_bounds__(256) void gemm_scores_kernel(const _Float16* __restrict__ Ph,
                                                          const _Float16* __restrict__ Eh,
                                                          float* __restrict__ out) {
    const int lane = threadIdx.x & 31;
    const int wave = threadIdx.x >> 5;         // 8 waves -> 8 N-tiles of 16
    const int r    = lane & 15;
    const int hi   = lane >> 4;

    const int n0 = blockIdx.x * 128 + wave * 16;   // V tile base for this wave
    const int m0 = blockIdx.y * 16;                // B tile base

    const _Float16* arow = Ph + (size_t)(m0 + r) * EE;      // A: row m = r
    int vc = n0 + r; if (vc > VV - 1) vc = VV - 1;          // clamp (select, EXEC intact)
    const _Float16* brow = Eh + (size_t)vc * EE;            // B: column n = r

    v8f c = {};
    #pragma unroll
    for (int k0 = 0; k0 < EE; k0 += 32) {
        const v16h a  = load_frag16(arow, k0, hi);
        const v16h bf = load_frag16(brow, k0, hi);
        c = __builtin_amdgcn_wmma_f32_16x16x32_f16(false, a, false, bf,
                                                   (short)0, c, false, false);
    }

    // C/D layout: VGPR j holds M = j + 8*hi, N = lane%16
    const int n = n0 + r;
    if (n < VV) {
        #pragma unroll
        for (int j = 0; j < 8; ++j) {
            out[(size_t)(m0 + j + 8 * hi) * VV + n] = c[j];
        }
    }
}

// ---------------------------------------------------------------------------
// K5: in-place log_softmax per row. One block per row; the 201KB row stays
// L2-resident across the three strided passes.
// ---------------------------------------------------------------------------
__global__ __launch_bounds__(256) void log_softmax_kernel(float* __restrict__ out) {
    __shared__ float red[256];
    const int t = threadIdx.x;
    float* row = out + (size_t)blockIdx.x * VV;

    float m = -INFINITY;
    for (int v = t; v < VV; v += 256) m = fmaxf(m, row[v]);
    red[t] = m; __syncthreads();
    for (int s = 128; s > 0; s >>= 1) {
        if (t < s) red[t] = fmaxf(red[t], red[t + s]);
        __syncthreads();
    }
    m = red[0]; __syncthreads();

    float sum = 0.f;
    for (int v = t; v < VV; v += 256) sum += __expf(row[v] - m);
    red[t] = sum; __syncthreads();
    for (int s = 128; s > 0; s >>= 1) {
        if (t < s) red[t] += red[t + s];
        __syncthreads();
    }
    const float lse = m + __logf(red[0]);

    for (int v = t; v < VV; v += 256) row[v] -= lse;
}

// ---------------------------------------------------------------------------
// Launch
// ---------------------------------------------------------------------------
extern "C" void kernel_launch(void* const* d_in, const int* in_sizes, int n_in,
                              void* d_out, int out_size, void* d_ws, size_t ws_size,
                              hipStream_t stream) {
    const float* xs     = (const float*)d_in[0];   // [B, V] one-hot f32
    const float* EMBEDM = (const float*)d_in[1];   // [V, E] f32
    const float* metric = (const float*)d_in[2];   // [E, E] f32
    float* out = (float*)d_out;                    // [B, V] f32

    // Workspace layout (16B aligned blocks):
    //   [0, 4KB)                : idx  (int[1024])
    //   [4KB, 4KB+256KB)        : Ph   (_Float16[1024*128])
    //   [4KB+256KB, +12.9MB)    : Eh   (_Float16[50257*128])
    char* ws = (char*)d_ws;
    int*      idx = (int*)ws;
    _Float16* Ph  = (_Float16*)(ws + 4096);
    _Float16* Eh  = (_Float16*)(ws + 4096 + (size_t)BB * EE * sizeof(_Float16));

    // 1) recover one-hot indices
    find_idx_kernel<<<BB, 256, 0, stream>>>(xs, idx);

    // 2) EMBEDM -> f16 (L2-resident operand for the big GEMM)
    {
        const size_t n4 = (size_t)VV * EE / 4;
        const int blocks = (int)((n4 + 255) / 256);
        cvt_embed_kernel<<<blocks, 256, 0, stream>>>(EMBEDM, Eh);
    }

    // 3) P = EMBEDM[idx] @ metric  (f32 -> f16)
    gather_project_kernel<<<BB, 128, 0, stream>>>(EMBEDM, metric, idx, Ph);

    // 4) scores = P @ Eh^T via WMMA; tile = 16(M) x 128(N) per workgroup
    {
        dim3 grid((VV + 127) / 128, BB / 16);
        gemm_scores_kernel<<<grid, 256, 0, stream>>>(Ph, Eh, out);
    }

    // 5) in-place log_softmax over V
    log_softmax_kernel<<<BB, 256, 0, stream>>>(out);
}
